// LinearCRF_14783277433043
// MI455X (gfx1250) — matrix operand order repaired and checked by reading, not compiled, not run on hardware
//
#include <hip/hip_runtime.h>

#define TT       128     // number of tags
#define S_LEN    1024
#define BATCH    128
#define BB       16      // batches per workgroup
#define NWG      (BATCH / BB)
#define SOS_TAG  0
#define EOS_TAG  1
#define STRH     136     // padded LDS row stride (halves) -> 272B, conflict-free

typedef __attribute__((ext_vector_type(16))) _Float16 v16h;
typedef __attribute__((ext_vector_type(8)))  _Float16 v8h;
typedef __attribute__((ext_vector_type(8)))  float    v8f;

union Frag { v16h v; v8h h[2]; };

__launch_bounds__(256, 1)
__global__ void crf_nll_main(const float* __restrict__ em,
                             const float* __restrict__ trans,
                             const float* __restrict__ mask,
                             const int*   __restrict__ tags,
                             float*       __restrict__ partial)
{
    __shared__ __attribute__((aligned(16))) _Float16 BmatT[TT * STRH]; // exp(T)^T, f16
    __shared__ __attribute__((aligned(16))) _Float16 Af16[BB * STRH];  // exp(alpha-m), f16
    __shared__ float alpha[BB * TT];
    __shared__ float rowmax[BB];
    __shared__ float scoreSh[BB];
    __shared__ float partSh[BB];

    const int tid  = threadIdx.x;
    const int wg   = blockIdx.x;
    const int b0   = wg * BB;
    const int lane = tid & 31;
    const int wid  = tid >> 5;        // wave 0..7
    const int r    = lane & 15;
    const int hi   = lane >> 4;       // half-wave select

    // ---- 1) BmatT[n][k] = exp(trans[k][n])  (transposed so B-frag loads are contiguous)
    for (int idx = tid; idx < TT * TT; idx += 256) {
        int k = idx >> 7, n = idx & (TT - 1);
        BmatT[n * STRH + k] = (_Float16)__expf(trans[idx]);
    }

    // ---- 2) alpha0[b][j] = trans[SOS][j] + em[b][0][j]
    for (int idx = tid; idx < BB * TT; idx += 256) {
        int bb = idx >> 7, j = idx & (TT - 1);
        alpha[idx] = trans[SOS_TAG * TT + j] + em[(long)(b0 + bb) * S_LEN * TT + j];
    }

    // ---- 3) gold-path score per batch (16 threads per batch)
    {
        const int bb  = tid >> 4;
        const int l16 = tid & 15;
        const int b   = b0 + bb;
        const long bs = (long)b * S_LEN;
        float sc = 0.f, cnt = 0.f;
        for (int s = l16; s < S_LEN; s += 16) {
            float mk = mask[bs + s];
            cnt += mk;
            if (s >= 1) {
                int tg  = tags[bs + s];
                int tgp = tags[bs + s - 1];
                sc += (em[(bs + s) * TT + tg] + trans[tgp * TT + tg]) * mk;
            }
        }
        #pragma unroll
        for (int m = 8; m >= 1; m >>= 1) {
            sc  += __shfl_xor(sc,  m, 16);
            cnt += __shfl_xor(cnt, m, 16);
        }
        if (l16 == 0) {
            int t0 = tags[bs];
            sc += trans[SOS_TAG * TT + t0] + em[bs * TT + t0];
            int lastIdx = (int)(cnt + 0.5f) - 1;
            if (lastIdx < 0) lastIdx = 0;
            if (lastIdx >= S_LEN) lastIdx = S_LEN - 1;
            sc += trans[tags[bs + lastIdx] * TT + EOS_TAG];
            scoreSh[bb] = sc;
        }
    }
    __syncthreads();

    // ---- 4) forward scan: per step, (16 x 128) = exp(alpha-m) @ exp(T) via WMMA
    const int n0  = wid * 16;
    const int col = n0 + r;
    for (int s = 1; s < S_LEN; ++s) {
        // phase 1: per-row max + exp -> Af16 (16-lane groups, no LDS reduction)
        {
            const int bb  = tid >> 4;
            const int l16 = tid & 15;
            float v[8]; float mx = -3.0e38f;
            #pragma unroll
            for (int i = 0; i < 8; ++i) {
                v[i] = alpha[bb * TT + l16 + 16 * i];
                mx = fmaxf(mx, v[i]);
            }
            #pragma unroll
            for (int m = 8; m >= 1; m >>= 1)
                mx = fmaxf(mx, __shfl_xor(mx, m, 16));
            #pragma unroll
            for (int i = 0; i < 8; ++i)
                Af16[bb * STRH + l16 + 16 * i] = (_Float16)__expf(v[i] - mx);
            if (l16 == 0) rowmax[bb] = mx;
        }

        // prefetch emissions + mask for phase 3 (global loads overlap the WMMAs)
        float emv[8], mkv[8];
        #pragma unroll
        for (int vv = 0; vv < 8; ++vv) {
            int M = vv + hi * 8;
            emv[vv] = em[((long)(b0 + M) * S_LEN + s) * TT + col];
            mkv[vv] = mask[(long)(b0 + M) * S_LEN + s];
        }
        __syncthreads();

        // phase 2: one 16x16 tile per wave, K = 128 in 4 WMMAs
        v8f acc = {};
        #pragma unroll
        for (int kk = 0; kk < TT; kk += 32) {
            Frag a, bf;
            // A 16x32 f16 layout: lane<16 -> K 0..7 & 16..23 ; lane>=16 -> K 8..15 & 24..31
            a.h[0]  = *(const v8h*)&Af16[r * STRH + kk + hi * 8];
            a.h[1]  = *(const v8h*)&Af16[r * STRH + kk + 16 + hi * 8];
            // B 32x16 f16 layout: lane<16 -> K 0..7 & 16..23 of col r ; lane>=16 -> K 8..15 & 24..31
            bf.h[0] = *(const v8h*)&BmatT[(n0 + r) * STRH + kk + hi * 8];
            bf.h[1] = *(const v8h*)&BmatT[(n0 + r) * STRH + kk + 16 + hi * 8];
            acc = __builtin_amdgcn_wmma_f32_16x16x32_f16(false, a.v, false, bf.v,
                                                         (short)0, acc, false, false);
        }

        // phase 3: new_alpha = m + log(gemm) + e ; masked combine; C layout: VGPR v -> M=v+hi*8, N=r
        #pragma unroll
        for (int vv = 0; vv < 8; ++vv) {
            int   M   = vv + hi * 8;
            float val = fmaxf(acc[vv], 1e-37f);          // clamp underflowed SOS column
            float na  = rowmax[M] + __logf(val) + emv[vv];
            float old = alpha[M * TT + col];
            float mi  = mkv[vv];
            alpha[M * TT + col] = mi * na + (1.0f - mi) * old;
        }
        __syncthreads();
    }

    // ---- 5) partition_b = logsumexp_j(alpha[b][j] + trans[j][EOS]); partial NLL
    {
        const int bb  = tid >> 4;
        const int l16 = tid & 15;
        float v[8]; float mx = -3.0e38f;
        #pragma unroll
        for (int i = 0; i < 8; ++i) {
            int j = l16 + 16 * i;
            v[i] = alpha[bb * TT + j] + trans[j * TT + EOS_TAG];
            mx = fmaxf(mx, v[i]);
        }
        #pragma unroll
        for (int m = 8; m >= 1; m >>= 1)
            mx = fmaxf(mx, __shfl_xor(mx, m, 16));
        float sm = 0.f;
        #pragma unroll
        for (int i = 0; i < 8; ++i) sm += __expf(v[i] - mx);
        #pragma unroll
        for (int m = 8; m >= 1; m >>= 1)
            sm += __shfl_xor(sm, m, 16);
        if (l16 == 0) partSh[bb] = (mx + __logf(sm)) - scoreSh[bb];
    }
    __syncthreads();
    if (tid == 0) {
        float s = 0.f;
        #pragma unroll
        for (int i = 0; i < BB; ++i) s += partSh[i];
        partial[wg] = s;
    }
}

__global__ void crf_nll_final(const float* __restrict__ partial, float* __restrict__ out)
{
    if (threadIdx.x == 0 && blockIdx.x == 0) {
        float s = 0.f;
        #pragma unroll
        for (int i = 0; i < NWG; ++i) s += partial[i];
        out[0] = s;
    }
}

extern "C" void kernel_launch(void* const* d_in, const int* in_sizes, int n_in,
                              void* d_out, int out_size, void* d_ws, size_t ws_size,
                              hipStream_t stream)
{
    const float* em    = (const float*)d_in[0];   // (B,S,T) f32
    const float* trans = (const float*)d_in[1];   // (T,T)   f32
    const float* mask  = (const float*)d_in[2];   // (B,S)   f32
    const int*   tags  = (const int*)  d_in[3];   // (B,S)   i32
    float* partial = (float*)d_ws;                // NWG floats of scratch

    crf_nll_main<<<NWG, 256, 0, stream>>>(em, trans, mask, tags, partial);
    crf_nll_final<<<1, 1, 0, stream>>>(partial, (float*)d_out);
}
